// CompressedCausalAttention_20486994002091
// MI455X (gfx1250) — matrix operand order repaired and checked by reading, not compiled, not run on hardware
//
#include <hip/hip_runtime.h>

typedef __attribute__((ext_vector_type(16))) _Float16 v16h;
typedef __attribute__((ext_vector_type(8)))  _Float16 v8h;
typedef __attribute__((ext_vector_type(4)))  _Float16 v4h;
typedef __attribute__((ext_vector_type(8)))  float    v8f;
typedef __attribute__((ext_vector_type(4)))  unsigned int u32x4;
typedef __attribute__((ext_vector_type(8)))  int i32x8;
typedef __attribute__((ext_vector_type(4)))  int i32x4;

union F16x16 { v16h v; v8h h[2]; };

static constexpr int S  = 1024;
static constexpr int B  = 4;
static constexpr int C  = 1024;
static constexpr int H  = 16;
static constexpr int CC = 64;
static constexpr int NTOK = S * B;   // 4096 token rows (s*B + b)

#if __has_builtin(__builtin_amdgcn_tensor_load_to_lds)
#define USE_TDM 1
#else
#define USE_TDM 0
#endif

#if USE_TDM
// LDS-resident volatile view: AS(3) pointers are 32-bit LDS byte offsets.
// volatile => loads are side effects (cannot fold to undef: the TDM's LDS
// writes are invisible to clang); AS(3) => real ds_load_b128, not flat.
typedef __attribute__((address_space(3))) const volatile v8h lds_cv8h;
#endif

// ---------------------------------------------------------------------------
// Kernel 0a: xh = f16(x + pe), vectorized 4-wide
// ---------------------------------------------------------------------------
__global__ __launch_bounds__(256) void prep_add_cvt(
    const float* __restrict__ x, const float* __restrict__ pe,
    _Float16* __restrict__ xh, int n4)
{
  int i = blockIdx.x * blockDim.x + threadIdx.x;
  if (i >= n4) return;
  float4 a = ((const float4*)x)[i];
  float4 b = ((const float4*)pe)[i];
  v4h o;
  o[0] = (_Float16)(a.x + b.x);
  o[1] = (_Float16)(a.y + b.y);
  o[2] = (_Float16)(a.z + b.z);
  o[3] = (_Float16)(a.w + b.w);
  ((v4h*)xh)[i] = o;
}

// ---------------------------------------------------------------------------
// Kernel 0b: one-shot f32 -> f16 weight conversion (weights reused 16-48x)
// ---------------------------------------------------------------------------
__global__ __launch_bounds__(256) void cvt_f32_f16(
    const float* __restrict__ w, _Float16* __restrict__ wh, int n4)
{
  int i = blockIdx.x * blockDim.x + threadIdx.x;
  if (i >= n4) return;
  float4 a = ((const float4*)w)[i];
  v4h o;
  o[0] = (_Float16)a.x; o[1] = (_Float16)a.y;
  o[2] = (_Float16)a.z; o[3] = (_Float16)a.w;
  ((v4h*)wh)[i] = o;
}

// ---------------------------------------------------------------------------
// TDM: program a 2D D# (data_size=2B, tile 32(K) x 64(N)) and kick the DMA.
// CDNA5 ISA: group0 = {flags, lds_addr, global_addr, type=2},
// group1 = {mask/data_size, dims, tile dims, strides}. LDS offset of the (only)
// __shared__ buffer is 0, so ldsOff is just the byte offset of the dest buffer.
// ---------------------------------------------------------------------------
#if USE_TDM
__device__ __forceinline__ void tdm_issue(const _Float16* W16, int K, int Ntot,
                                          int n0g, int k0, unsigned ldsOff)
{
  unsigned long long ga =
      (unsigned long long)(size_t)(W16 + (size_t)n0g * K + k0);
  u32x4 g0;
  g0[0] = 1u;                                          // count=1, normal mode
  g0[1] = ldsOff;                                      // lds_addr (bytes)
  g0[2] = (unsigned)(ga & 0xffffffffull);              // global_addr[31:0]
  g0[3] = (unsigned)((ga >> 32) & 0x01ffffffull)       // global_addr[56:32]
          | (2u << 30);                                // type = 2 ("image")
  i32x8 g1;
  g1[0] = (int)(1u << 16);                             // data_size = 2 bytes
  g1[1] = (int)(((unsigned)K & 0xffffu) << 16);        // tensor_dim0[15:0]
  g1[2] = (int)(((unsigned)K >> 16) |
                (((unsigned)Ntot & 0xffffu) << 16));   // dim0 hi | dim1 lo
  g1[3] = (int)(((unsigned)Ntot >> 16) | (32u << 16)); // dim1 hi | tile_dim0=32
  g1[4] = 64;                                          // tile_dim1=64, dim2=0
  g1[5] = K;                                           // dim0_stride[31:0]
  g1[6] = 0;
  g1[7] = 0;
  i32x4 gz; gz[0] = gz[1] = gz[2] = gz[3] = 0;
#if __clang_major__ >= 23
  i32x8 gz8;
#pragma unroll
  for (int j = 0; j < 8; ++j) gz8[j] = 0;
  __builtin_amdgcn_tensor_load_to_lds(g0, g1, gz, gz, gz8, 0);
#else
  __builtin_amdgcn_tensor_load_to_lds(g0, g1, gz, gz, 0);
#endif
}
#endif

// ---------------------------------------------------------------------------
// WMMA GEMM mainloop: 8 waves/WG, WG tile 128(M) x 64(N), K step 32.
// A: f16 [M][K] row-major. W16: f16 [N][K] row-major. Weight tiles staged into
// double-buffered LDS by the Tensor Data Mover (wave 0 drives, TENSORcnt
// tracked), overlapping DMA of tile t+1 with WMMAs on tile t.
// ---------------------------------------------------------------------------
__device__ __forceinline__ void gemm_mainloop(
    const _Float16* __restrict__ A, const _Float16* __restrict__ W16,
    int K, int Ntot, int mrow, int n0g, _Float16* ldsb, int tid, v8f acc[4])
{
  const int lane = tid & 31;
  const int hx   = lane >> 4;   // lane half (0/1)
  const int lo   = lane & 15;
  const int nIter = K / 32;

#if USE_TDM
  if (tid < 32) tdm_issue(W16, K, Ntot, n0g, 0, 0u);
  asm volatile("" ::: "memory");         // DMA writes LDS behind clang's back
#endif
  for (int it = 0; it < nIter; ++it) {
    const int k0 = it * 32;
    __syncthreads();                     // buffer being refilled is consumed
#if USE_TDM
    if (tid < 32) {
      if (it + 1 < nIter) {
        tdm_issue(W16, K, Ntot, n0g, k0 + 32, (unsigned)(((it + 1) & 1) * 4096));
        __builtin_amdgcn_s_wait_tensorcnt(1);   // tile it done, it+1 in flight
      } else {
        __builtin_amdgcn_s_wait_tensorcnt(0);
      }
    }
#else
    {
      // cooperative fallback: copy W16 tile [64n x 32k] into buffer it&1
      int n  = tid >> 2;
      int kc = (tid & 3) << 3;
      *(v8h*)(ldsb + ((it & 1) * 2048) + n * 32 + kc) =
          *(const v8h*)(W16 + (size_t)(n0g + n) * K + k0 + kc);
    }
#endif
    __syncthreads();                     // tile it visible to all waves
#if USE_TDM
    asm volatile("" ::: "memory");       // LDS contents changed by the TDM
#endif

    // A fragment: lane holds row lo; K chunk hx*16..+15
    F16x16 a;
    const _Float16* ap = A + (size_t)mrow * K + k0 + hx * 16;
    a.h[0] = *(const v8h*)ap;
    a.h[1] = *(const v8h*)(ap + 8);
    if (k0 + 32 < K)
      __builtin_prefetch(A + (size_t)mrow * K + k0 + 32, 0, 1);

    // load all 4 B fragments first, then issue WMMAs back-to-back
    F16x16 bf[4];
#if USE_TDM
    const unsigned bufBase = (unsigned)((it & 1) * 4096);
#pragma unroll
    for (int i = 0; i < 4; ++i) {
      const unsigned boff = bufBase + (unsigned)(((i * 16 + lo) * 32 + hx * 16) * 2);
      lds_cv8h* bp = (lds_cv8h*)boff;          // inttoptr: LDS byte offset
      bf[i].h[0] = bp[0];
      bf[i].h[1] = bp[1];
    }
#else
    const _Float16* lb = ldsb + (it & 1) * 2048;
#pragma unroll
    for (int i = 0; i < 4; ++i) {
      const _Float16* bp = lb + (i * 16 + lo) * 32 + hx * 16;
      bf[i].h[0] = *(const v8h*)bp;
      bf[i].h[1] = *(const v8h*)(bp + 8);
    }
#endif
#pragma unroll
    for (int i = 0; i < 4; ++i)
      acc[i] = __builtin_amdgcn_wmma_f32_16x16x32_f16(
          false, a.v, false, bf[i].v, (short)0, acc[i], false, false);
  }
}

// ---------------------------------------------------------------------------
// Kernel 1: QKV projection. out cols: [0,1024)=Q, [1024,2048)=K, [2048,3072)=V
// Q,K stored [B*H][S][CC] f16; V stored transposed [B*H][CC][S] f16.
// ---------------------------------------------------------------------------
__global__ __launch_bounds__(256) void qkv_gemm_kernel(
    const _Float16* __restrict__ xh, const _Float16* __restrict__ Wq16,
    const float* __restrict__ bqkv,
    _Float16* __restrict__ qb, _Float16* __restrict__ kb,
    _Float16* __restrict__ vt)
{
  __shared__ _Float16 ldsb[2 * 64 * 32];
  const int tid  = threadIdx.x;
  const int wave = tid >> 5;
  const int lane = tid & 31;
  const int hx   = lane >> 4;
  const int lo   = lane & 15;
  const int mrow = blockIdx.x * 128 + wave * 16 + lo;
  const int n0g  = blockIdx.y * 64;

#if USE_TDM
  // never-taken, non-foldable store: keeps ldsb allocated (8KB group segment)
  // even though the TDM path reads it only via inttoptr AS(3) pointers.
  if (__builtin_expect((int)blockIdx.x == -1, 0)) ldsb[tid] = (_Float16)0;
#endif

  v8f acc[4];
#pragma unroll
  for (int i = 0; i < 4; ++i)
#pragma unroll
    for (int j = 0; j < 8; ++j) acc[i][j] = 0.f;

  gemm_mainloop(xh, Wq16, C, 3 * C, mrow, n0g, ldsb, tid, acc);

#pragma unroll
  for (int i = 0; i < 4; ++i) {
    const int col  = n0g + i * 16 + lo;
    const float bias = bqkv[col];
    const int seg = col >> 10;       // 0=q 1=k 2=v
    const int cl  = col & 1023;
    const int hh  = cl >> 6;
    const int cc  = cl & 63;
#pragma unroll
    for (int r = 0; r < 8; ++r) {
      const int row = blockIdx.x * 128 + wave * 16 + r + 8 * hx;
      const int s_  = row >> 2;      // row = s*B + b
      const int bb  = row & 3;
      const size_t bh = (size_t)bb * H + hh;
      const _Float16 hv = (_Float16)(acc[i][r] + bias);
      if (seg == 0)      qb[(bh * S + s_) * CC + cc] = hv;
      else if (seg == 1) kb[(bh * S + s_) * CC + cc] = hv;
      else               vt[(bh * CC + cc) * S + s_] = hv;
    }
  }
}

// ---------------------------------------------------------------------------
// Kernel 2: flash attention. grid = (B*H, S/128); 8 waves/WG, 16 queries/wave.
// Online softmax over 32-key blocks; causal + key padding masks.
// ---------------------------------------------------------------------------
__global__ __launch_bounds__(256) void attn_kernel(
    const _Float16* __restrict__ qbuf, const _Float16* __restrict__ kbuf,
    const _Float16* __restrict__ vT, const unsigned char* __restrict__ pad,
    _Float16* __restrict__ attnb)
{
  __shared__ _Float16 lds_p[8][16][32];   // per-wave P patch (C/D -> A relayout)
  const int tid  = threadIdx.x;
  const int wave = tid >> 5;
  const int lane = tid & 31;
  const int hx   = lane >> 4;
  const int lo   = lane & 15;
  const int bh   = blockIdx.x;            // b*H + h
  const int b    = bh >> 4;
  const int h    = bh & 15;
  const int qbase = blockIdx.y * 128 + wave * 16;

  const _Float16* Q  = qbuf + (size_t)bh * S * CC;
  const _Float16* Kp = kbuf + (size_t)bh * S * CC;
  const _Float16* Vt = vT   + (size_t)bh * CC * S;

  // Q fragments, feature chunks c*32 (A-layout)
  F16x16 qa[2];
#pragma unroll
  for (int c = 0; c < 2; ++c) {
    const _Float16* qp = Q + (size_t)(qbase + lo) * CC + c * 32 + hx * 16;
    qa[c].h[0] = *(const v8h*)qp;
    qa[c].h[1] = *(const v8h*)(qp + 8);
  }

  v8f o[4];
#pragma unroll
  for (int t = 0; t < 4; ++t)
#pragma unroll
    for (int j = 0; j < 8; ++j) o[t][j] = 0.f;
  float mrow[8], lrow[8];
#pragma unroll
  for (int r = 0; r < 8; ++r) { mrow[r] = -1e30f; lrow[r] = 0.f; }

  v8f z;
#pragma unroll
  for (int j = 0; j < 8; ++j) z[j] = 0.f;

  const int nblk = (qbase + 47) >> 5;     // key blocks covering causal range
  for (int t = 0; t < nblk; ++t) {
    const int tb = t * 32;
    // ---- scores: 2 key subtiles x 2 feature chunks; batch loads, then WMMAs
    v8f s0 = z, s1 = z;
    {
      F16x16 kf[4];
      const _Float16* kp0 = Kp + (size_t)(tb + lo) * CC + hx * 16;
      const _Float16* kp1 = Kp + (size_t)(tb + 16 + lo) * CC + hx * 16;
      kf[0].h[0] = *(const v8h*)kp0;        kf[0].h[1] = *(const v8h*)(kp0 + 8);
      kf[1].h[0] = *(const v8h*)(kp0 + 32); kf[1].h[1] = *(const v8h*)(kp0 + 40);
      kf[2].h[0] = *(const v8h*)kp1;        kf[2].h[1] = *(const v8h*)(kp1 + 8);
      kf[3].h[0] = *(const v8h*)(kp1 + 32); kf[3].h[1] = *(const v8h*)(kp1 + 40);
      s0 = __builtin_amdgcn_wmma_f32_16x16x32_f16(false, qa[0].v, false, kf[0].v,
                                                  (short)0, s0, false, false);
      s1 = __builtin_amdgcn_wmma_f32_16x16x32_f16(false, qa[0].v, false, kf[2].v,
                                                  (short)0, s1, false, false);
      s0 = __builtin_amdgcn_wmma_f32_16x16x32_f16(false, qa[1].v, false, kf[1].v,
                                                  (short)0, s0, false, false);
      s1 = __builtin_amdgcn_wmma_f32_16x16x32_f16(false, qa[1].v, false, kf[3].v,
                                                  (short)0, s1, false, false);
    }
    // ---- mask + online softmax --------------------------------------------
    const unsigned char pd0 = pad[(size_t)(tb + lo) * B + b];
    const unsigned char pd1 = pad[(size_t)(tb + 16 + lo) * B + b];
    const float NEG = -__builtin_inff();
#pragma unroll
    for (int r = 0; r < 8; ++r) {
      const int qi = qbase + r + 8 * hx;
      float v0 = s0[r] * 0.125f;
      float v1 = s1[r] * 0.125f;
      if ((tb + lo) > qi || pd0)      v0 = NEG;
      if ((tb + 16 + lo) > qi || pd1) v1 = NEG;
      float mx = fmaxf(v0, v1);
      mx = fmaxf(mx, __shfl_xor(mx, 1, 16));
      mx = fmaxf(mx, __shfl_xor(mx, 2, 16));
      mx = fmaxf(mx, __shfl_xor(mx, 4, 16));
      mx = fmaxf(mx, __shfl_xor(mx, 8, 16));
      const float mnew = fmaxf(mrow[r], mx);
      const float al   = __expf(mrow[r] - mnew);
      const float p0   = __expf(v0 - mnew);
      const float p1   = __expf(v1 - mnew);
      float rs = p0 + p1;
      rs += __shfl_xor(rs, 1, 16);
      rs += __shfl_xor(rs, 2, 16);
      rs += __shfl_xor(rs, 4, 16);
      rs += __shfl_xor(rs, 8, 16);
      lrow[r] = lrow[r] * al + rs;
      mrow[r] = mnew;
      o[0][r] *= al; o[1][r] *= al; o[2][r] *= al; o[3][r] *= al;
      lds_p[wave][r + 8 * hx][lo]      = (_Float16)p0;
      lds_p[wave][r + 8 * hx][16 + lo] = (_Float16)p1;
    }
    asm volatile("s_wait_dscnt 0x0" ::: "memory");
    // P back as A-matrix fragment (16 queries x 32 keys)
    F16x16 pa;
    pa.h[0] = *(const v8h*)&lds_p[wave][lo][hx * 16];
    pa.h[1] = *(const v8h*)&lds_p[wave][lo][hx * 16 + 8];
    // ---- O += P * V^T over 4 feature subtiles (batch loads, then WMMAs) ---
    F16x16 vf[4];
#pragma unroll
    for (int ft = 0; ft < 4; ++ft) {
      const _Float16* vp = Vt + (size_t)(ft * 16 + lo) * S + tb + hx * 16;
      vf[ft].h[0] = *(const v8h*)vp;
      vf[ft].h[1] = *(const v8h*)(vp + 8);
    }
#pragma unroll
    for (int ft = 0; ft < 4; ++ft)
      o[ft] = __builtin_amdgcn_wmma_f32_16x16x32_f16(false, pa.v, false, vf[ft].v,
                                                     (short)0, o[ft], false, false);
  }
  // ---- epilogue: normalize and store f16 [token(s*B+b)][h*64+f] ------------
#pragma unroll
  for (int r = 0; r < 8; ++r) {
    const float inv = lrow[r] > 0.f ? 1.f / lrow[r] : 0.f;
    const int qi = qbase + r + 8 * hx;
    const size_t rowb = (size_t)qi * B + b;
#pragma unroll
    for (int ft = 0; ft < 4; ++ft) {
      const int col = h * CC + ft * 16 + lo;
      attnb[rowb * C + col] = (_Float16)(o[ft][r] * inv);
    }
  }
}

// ---------------------------------------------------------------------------
// Kernel 3: output projection, f32 epilogue straight to d_out
// ---------------------------------------------------------------------------
__global__ __launch_bounds__(256) void out_gemm_kernel(
    const _Float16* __restrict__ attnb, const _Float16* __restrict__ Wo16,
    const float* __restrict__ bo, float* __restrict__ out)
{
  __shared__ _Float16 ldsb[2 * 64 * 32];
  const int tid  = threadIdx.x;
  const int wave = tid >> 5;
  const int lane = tid & 31;
  const int hx   = lane >> 4;
  const int lo   = lane & 15;
  const int mrow = blockIdx.x * 128 + wave * 16 + lo;
  const int n0g  = blockIdx.y * 64;

#if USE_TDM
  if (__builtin_expect((int)blockIdx.x == -1, 0)) ldsb[tid] = (_Float16)0;
#endif

  v8f acc[4];
#pragma unroll
  for (int i = 0; i < 4; ++i)
#pragma unroll
    for (int j = 0; j < 8; ++j) acc[i][j] = 0.f;

  gemm_mainloop(attnb, Wo16, C, C, mrow, n0g, ldsb, tid, acc);

#pragma unroll
  for (int i = 0; i < 4; ++i) {
    const int col = n0g + i * 16 + lo;
    const float bias = bo[col];
#pragma unroll
    for (int r = 0; r < 8; ++r) {
      const int row = blockIdx.x * 128 + wave * 16 + r + 8 * hx;
      out[(size_t)row * C + col] = acc[i][r] + bias;
    }
  }
}

// ---------------------------------------------------------------------------
extern "C" void kernel_launch(void* const* d_in, const int* in_sizes, int n_in,
                              void* d_out, int out_size, void* d_ws, size_t ws_size,
                              hipStream_t stream)
{
  const float* x    = (const float*)d_in[0];
  const float* pe   = (const float*)d_in[1];
  // d_in[2] = content_mask: causal, computed analytically -> never read
  const unsigned char* pad = (const unsigned char*)d_in[3];
  const float* Wqkv = (const float*)d_in[4];
  const float* bqkv = (const float*)d_in[5];
  const float* Wo   = (const float*)d_in[6];
  const float* bo   = (const float*)d_in[7];
  float* out = (float*)d_out;

  char* ws = (char*)d_ws;
  _Float16* xh   = (_Float16*)(ws);               // 8 MB (reused as attn out)
  _Float16* qb   = (_Float16*)(ws + (8  << 20));  // 8 MB
  _Float16* kb   = (_Float16*)(ws + (16 << 20));  // 8 MB
  _Float16* vt   = (_Float16*)(ws + (24 << 20));  // 8 MB (V transposed)
  _Float16* wq16 = (_Float16*)(ws + (32 << 20));  // 6 MB (Wqkv f16)
  _Float16* wo16 = (_Float16*)(ws + (38 << 20));  // 2 MB (Wo f16)
  _Float16* attnb = xh;                           // alias: xh dead post-QKV

  const int n4x = NTOK * C / 4;                   // 1,048,576 float4 groups
  prep_add_cvt<<<dim3(n4x / 256), dim3(256), 0, stream>>>(x, pe, xh, n4x);
  const int n4wq = 3 * C * C / 4;                 // 786,432
  cvt_f32_f16<<<dim3(n4wq / 256), dim3(256), 0, stream>>>(Wqkv, wq16, n4wq);
  const int n4wo = C * C / 4;                     // 262,144
  cvt_f32_f16<<<dim3(n4wo / 256), dim3(256), 0, stream>>>(Wo, wo16, n4wo);

  qkv_gemm_kernel<<<dim3(NTOK / 128, (3 * C) / 64), dim3(256), 0, stream>>>(
      xh, wq16, bqkv, qb, kb, vt);
  attn_kernel<<<dim3(B * H, S / 128), dim3(256), 0, stream>>>(
      qb, kb, vt, pad, attnb);
  out_gemm_kernel<<<dim3(NTOK / 128, C / 64), dim3(256), 0, stream>>>(
      attnb, wo16, bo, out);
}
